// APPNP_Net_27908697489843
// MI455X (gfx1250) — matrix operand order repaired and checked by reading, not compiled, not run on hardware
//
#include <hip/hip_runtime.h>
#include <hip/hip_bf16.h>

#define NND 100000
#define NFEAT 512
#define HID 64
#define NCLS 40
#define NEDGE 3200000
#define KPROP 10
#define ALPHA 0.1f

typedef __attribute__((ext_vector_type(16))) _Float16 v16h;
typedef __attribute__((ext_vector_type(8)))  float    v8f;

// ---------------- weight prepack: f32 row-major -> WMMA-B fragment-ordered f16 ----
// B frag (16-bit, 32xK=..x16): lane<16 holds K=0..7,16..23 of col N=lane; lane>=16 holds
// K=8..15,24..31 of col N=lane-16; 16 sequential halves per lane.
__global__ void pack_w1(const float* __restrict__ W1, _Float16* __restrict__ out) {
    int p = blockIdx.x * 256 + threadIdx.x;          // 16 kb * 4 ntile * 32 lane * 16
    int kbIdx = p >> 11;
    int rem   = p & 2047;
    int ntile = rem >> 9;
    int rem2  = rem & 511;
    int lane  = rem2 >> 4;
    int i     = rem2 & 15;
    int k = kbIdx * 32 + ((lane >> 4) << 3) + (i & 7) + ((i >> 3) << 4);
    int n = ntile * 16 + (lane & 15);
    out[p] = (_Float16)W1[k * HID + n];
}

__global__ void pack_w2(const float* __restrict__ W2, _Float16* __restrict__ out) {
    int p = blockIdx.x * 256 + threadIdx.x;          // 2 kb * 3 ntile * 32 lane * 16
    int kbIdx = p / 1536;
    int rem   = p % 1536;
    int ntile = rem >> 9;
    int rem2  = rem & 511;
    int lane  = rem2 >> 4;
    int i     = rem2 & 15;
    int k   = kbIdx * 32 + ((lane >> 4) << 3) + (i & 7) + ((i >> 3) << 4);
    int col = ntile * 16 + (lane & 15);
    out[p] = (col < NCLS) ? (_Float16)W2[k * NCLS + col] : (_Float16)0.0f;
}

// ---------------- GEMM1: h1 = relu(x @ W1 + b1), [N,64] --------------------------
// 256 thr = 8 waves; wave -> (msub 0/1, ntile 0..3); block covers 32 rows.
__global__ void gemm1_relu(const float* __restrict__ x, const _Float16* __restrict__ w1p,
                           const float* __restrict__ b1, float* __restrict__ h1) {
    int wave  = threadIdx.x >> 5;
    int lane  = threadIdx.x & 31;
    int msub  = wave >> 2;
    int ntile = wave & 3;
    int row   = blockIdx.x * 32 + msub * 16 + (lane & 15);
    const float* arow = x + (size_t)row * NFEAT + ((lane >> 4) << 3);
    const v16h* bfrag = (const v16h*)w1p;

    v8f acc = {};
    for (int kb = 0; kb < NFEAT / 32; ++kb) {
        v16h a, b;
        const float* ap = arow + kb * 32;
        #pragma unroll
        for (int i = 0; i < 8; ++i) {            // A layout: K 0-7/8-15 then 16-23/24-31
            a[i]     = (_Float16)ap[i];
            a[8 + i] = (_Float16)ap[16 + i];
        }
        b = bfrag[(kb * 4 + ntile) * 32 + lane];
        acc = __builtin_amdgcn_wmma_f32_16x16x32_f16(false, a, false, b,
                                                     (short)0, acc, false, false);
    }
    int col  = ntile * 16 + (lane & 15);
    float bv = b1[col];
    int m0   = blockIdx.x * 32 + msub * 16 + ((lane >> 4) << 3);
    float* o = h1 + (size_t)m0 * HID + col;
    #pragma unroll
    for (int r = 0; r < 8; ++r) {
        float v = acc[r] + bv;
        o[(size_t)r * HID] = v > 0.0f ? v : 0.0f;
    }
}

// ---------------- GEMM2: h0 = h1 @ W2 + b2, [N,40] (cols padded to 48) -----------
// 192 thr = 6 waves; wave -> (msub 0/1, ntile 0..2); block covers 32 rows.
__global__ void gemm2(const float* __restrict__ h1, const _Float16* __restrict__ w2p,
                      const float* __restrict__ b2, float* __restrict__ h0) {
    int wave  = threadIdx.x >> 5;
    int lane  = threadIdx.x & 31;
    int msub  = wave / 3;
    int ntile = wave % 3;
    int row   = blockIdx.x * 32 + msub * 16 + (lane & 15);
    const float* arow = h1 + (size_t)row * HID + ((lane >> 4) << 3);
    const v16h* bfrag = (const v16h*)w2p;

    v8f acc = {};
    #pragma unroll
    for (int kb = 0; kb < 2; ++kb) {
        v16h a, b;
        const float* ap = arow + kb * 32;
        #pragma unroll
        for (int i = 0; i < 8; ++i) {
            a[i]     = (_Float16)ap[i];
            a[8 + i] = (_Float16)ap[16 + i];
        }
        b = bfrag[(kb * 3 + ntile) * 32 + lane];
        acc = __builtin_amdgcn_wmma_f32_16x16x32_f16(false, a, false, b,
                                                     (short)0, acc, false, false);
    }
    int col = ntile * 16 + (lane & 15);
    if (col < NCLS) {                             // branch after WMMA: EXEC stays full
        float bv = b2[col];
        int m0   = blockIdx.x * 32 + msub * 16 + ((lane >> 4) << 3);
        float* o = h0 + (size_t)m0 * NCLS + col;
        #pragma unroll
        for (int r = 0; r < 8; ++r) o[(size_t)r * NCLS] = acc[r] + bv;
    }
}

// ---------------- degree / dinv --------------------------------------------------
__global__ void deg_init(float* __restrict__ deg) {
    int i = blockIdx.x * 256 + threadIdx.x;
    if (i < NND) deg[i] = 1.0f;                   // self-loop
}
__global__ void deg_acc(const int* __restrict__ col, float* __restrict__ deg) {
    int e = blockIdx.x * 256 + threadIdx.x;       // NEDGE exact
    unsafeAtomicAdd(&deg[col[e]], 1.0f);
}
__global__ void deg_rsqrt(float* __restrict__ deg) {
    int i = blockIdx.x * 256 + threadIdx.x;
    if (i < NND) deg[i] = rsqrtf(deg[i]);
}

// ---------------- propagation: z' = a*h0 + (1-a)*(self + edges) ------------------
__global__ void prop_init(const float* __restrict__ h0, const float* __restrict__ zin,
                          const float* __restrict__ dinv, float* __restrict__ znext) {
    int idx  = blockIdx.x * 256 + threadIdx.x;    // NND*NCLS exact
    int node = idx / NCLS;
    float di = dinv[node];
    znext[idx] = ALPHA * h0[idx] + (1.0f - ALPHA) * di * di * zin[idx];
}

// one wave per edge; 32 lanes cover 40 channels (lanes 0..7 do a second one)
__global__ void prop_edges(const int* __restrict__ row, const int* __restrict__ col,
                           const float* __restrict__ dinv, const float* __restrict__ zin,
                           float* __restrict__ znext) {
    int e    = blockIdx.x * 8 + (threadIdx.x >> 5);   // NEDGE exact
    int lane = threadIdx.x & 31;
    int r = row[e];
    int t = col[e];
    float w = (1.0f - ALPHA) * dinv[r] * dinv[t];
    const float* zr = zin   + (size_t)r * NCLS;
    float*       zt = znext + (size_t)t * NCLS;
    unsafeAtomicAdd(&zt[lane], w * zr[lane]);
    if (lane < 8) unsafeAtomicAdd(&zt[32 + lane], w * zr[32 + lane]);
}

// ---------------- log-softmax over 40 classes, one wave per row ------------------
__global__ void logsoftmax40(const float* __restrict__ z, float* __restrict__ out) {
    int row  = blockIdx.x * 8 + (threadIdx.x >> 5);   // NND/8 exact
    int lane = threadIdx.x & 31;
    const float* zr = z + (size_t)row * NCLS;
    float v0 = zr[lane];
    float v1 = (lane < 8) ? zr[32 + lane] : -3.402823466e38f;
    float m = fmaxf(v0, v1);
    #pragma unroll
    for (int off = 16; off > 0; off >>= 1) m = fmaxf(m, __shfl_xor(m, off, 32));
    float s = __expf(v0 - m) + ((lane < 8) ? __expf(v1 - m) : 0.0f);
    #pragma unroll
    for (int off = 16; off > 0; off >>= 1) s += __shfl_xor(s, off, 32);
    float lse = m + __logf(s);
    float* o = out + (size_t)row * NCLS;
    o[lane] = v0 - lse;
    if (lane < 8) o[32 + lane] = v1 - lse;
}

extern "C" void kernel_launch(void* const* d_in, const int* in_sizes, int n_in,
                              void* d_out, int out_size, void* d_ws, size_t ws_size,
                              hipStream_t stream) {
    const float* x  = (const float*)d_in[0];
    const float* W1 = (const float*)d_in[1];
    const float* b1 = (const float*)d_in[2];
    const float* W2 = (const float*)d_in[3];
    const float* b2 = (const float*)d_in[4];
    const int*   er = (const int*)d_in[5];          // edge_index[0] (source/gather)
    const int*   ec = er + NEDGE;                   // edge_index[1] (target/scatter)

    char* ws = (char*)d_ws;                         // all offsets 256B aligned
    float*    h1   = (float*)(ws + 0);              // 25,600,000 B
    float*    h0   = (float*)(ws + 25600000);       // 16,000,000 B
    float*    zA   = (float*)(ws + 41600000);       // 16,000,000 B
    float*    zB   = (float*)(ws + 57600000);       // 16,000,000 B
    float*    dinv = (float*)(ws + 73600000);       //    400,000 B
    _Float16* w1p  = (_Float16*)(ws + 74000128);    //     65,536 B
    _Float16* w2p  = (_Float16*)(ws + 74065920);    //      6,144 B

    pack_w1<<<128, 256, 0, stream>>>(W1, w1p);
    pack_w2<<<12,  256, 0, stream>>>(W2, w2p);

    gemm1_relu<<<NND / 32, 256, 0, stream>>>(x, w1p, b1, h1);
    gemm2     <<<NND / 32, 192, 0, stream>>>(h1, w2p, b2, h0);

    deg_init <<<(NND + 255) / 256, 256, 0, stream>>>(dinv);
    deg_acc  <<<NEDGE / 256,       256, 0, stream>>>(ec, dinv);
    deg_rsqrt<<<(NND + 255) / 256, 256, 0, stream>>>(dinv);

    const float* zin = h0;
    for (int k = 0; k < KPROP; ++k) {
        float* znext = (k & 1) ? zB : zA;
        prop_init <<<(NND * NCLS) / 256, 256, 0, stream>>>(h0, zin, dinv, znext);
        prop_edges<<<NEDGE / 8,          256, 0, stream>>>(er, ec, dinv, zin, znext);
        zin = znext;
    }

    logsoftmax40<<<NND / 8, 256, 0, stream>>>(zin, (float*)d_out);
}
